// SelfCF_HE_50560355008860
// MI455X (gfx1250) — matrix operand order repaired and checked by reading, not compiled, not run on hardware
//
#include <hip/hip_runtime.h>
#include <hip/hip_bf16.h>

typedef _Float16 v16h __attribute__((ext_vector_type(16)));
typedef float    v8f  __attribute__((ext_vector_type(8)));

#define DD 64          // embedding dim (fixed by reference)
#define LAYERS 3
#define MOM 0.05f

// ---------------------------------------------------------------------------
// ego = concat(user_emb, item_emb); acc = ego; cur = ego   (float4 vectorized)
// ---------------------------------------------------------------------------
__global__ void init_concat_kernel(const float* __restrict__ ue,
                                   const float* __restrict__ ie,
                                   float* __restrict__ acc,
                                   float* __restrict__ cur,
                                   int uElems, int tot4) {
  int i = blockIdx.x * blockDim.x + threadIdx.x;
  if (i >= tot4) return;
  int u4 = uElems >> 2;
  float4 v = (i < u4) ? ((const float4*)ue)[i] : ((const float4*)ie)[i - u4];
  ((float4*)acc)[i] = v;
  ((float4*)cur)[i] = v;
}

__global__ void zero_kernel(float* __restrict__ p, int n4) {
  int i = blockIdx.x * blockDim.x + threadIdx.x;
  if (i >= n4) return;
  float4 z = {0.f, 0.f, 0.f, 0.f};
  ((float4*)p)[i] = z;
}

__global__ void accum_kernel(float* __restrict__ acc, const float* __restrict__ nxt, int n4) {
  int i = blockIdx.x * blockDim.x + threadIdx.x;
  if (i >= n4) return;
  float4 a = ((const float4*)acc)[i];
  float4 b = ((const float4*)nxt)[i];
  a.x += b.x; a.y += b.y; a.z += b.z; a.w += b.w;
  ((float4*)acc)[i] = a;
}

// ---------------------------------------------------------------------------
// SpMM scatter-add: y[row] += val * x[col].  One wave32 per nnz entry:
// each lane moves a float2 of the 64-float row (coalesced 256B line),
// scatter via global_atomic_add_f32. Dense buffers are L2-resident (38.4MB
// each << 192MB L2), so atomics resolve on-die.
// ---------------------------------------------------------------------------
__global__ void spmm_atomic_kernel(const int*   __restrict__ rows,
                                   const int*   __restrict__ cols,
                                   const float* __restrict__ vals,
                                   const float* __restrict__ x,
                                   float*       __restrict__ y,
                                   int nnz) {
  int lane   = threadIdx.x & 31;
  int wpb    = blockDim.x >> 5;
  int wave   = blockIdx.x * wpb + (threadIdx.x >> 5);
  int nwaves = gridDim.x * wpb;
  for (int e = wave; e < nnz; e += nwaves) {
    int   r = rows[e];
    int   c = cols[e];
    float v = vals[e];
    float2 xv = ((const float2*)(x + (size_t)c * DD))[lane];
    float* dst = y + (size_t)r * DD + lane * 2;
    atomicAdd(dst,     v * xv.x);
    atomicAdd(dst + 1, v * xv.y);
  }
}

// ---------------------------------------------------------------------------
// Gather online embeddings (acc/4), write momentum targets straight to d_out,
// stage u_online / i_online in workspace for the WMMA projection.
// out layout: [p_u | u_target | p_i | i_target], each B*64 floats.
// ---------------------------------------------------------------------------
__global__ void gather_target_kernel(const float* __restrict__ acc,
                                     const int*   __restrict__ users,
                                     const int*   __restrict__ items,
                                     const float* __restrict__ u_his,
                                     const float* __restrict__ i_his,
                                     float* __restrict__ u_on,
                                     float* __restrict__ i_on,
                                     float* __restrict__ out,
                                     int B, int U) {
  int t  = blockIdx.x * blockDim.x + threadIdx.x;
  int BD = B * DD;
  if (t >= 2 * BD) return;
  const float inv = 1.0f / (LAYERS + 1);
  if (t < BD) {
    int b = t >> 6, d = t & 63;
    int row = users[b];
    float v = acc[(size_t)row * DD + d] * inv;
    u_on[t] = v;
    out[BD + t] = u_his[(size_t)row * DD + d] * MOM + v * (1.0f - MOM);
  } else {
    int tt = t - BD;
    int b = tt >> 6, d = tt & 63;
    int row = items[b];
    float v = acc[(size_t)(U + row) * DD + d] * inv;
    i_on[tt] = v;
    out[3 * BD + tt] = i_his[(size_t)row * DD + d] * MOM + v * (1.0f - MOM);
  }
}

// ---------------------------------------------------------------------------
// P = X @ W^T + bias  via v_wmma_f32_16x16x32_f16.
// X: (B x 64) row-major f32, W: (64 x 64) row-major f32, P: (B x 64).
// One wave per 16x16 output tile; K=64 -> two WMMA steps.
// Fragment packing used here (wave32; contiguous k, same order for A and B):
//   A (16x32):  m = lane&15; lanes<16 hold K 0..15, lanes>=16 hold
//               K 16..31  ->  k = hi*16+j
//   B (32x16):  n = lane&15; lanes<16 hold K 0..15, lanes>=16 hold K 16..31
//   C/D (16x16 f32): col = lane&15, row r in VGPR r (+8 for upper lane half)
// ---------------------------------------------------------------------------
__global__ void __launch_bounds__(256)
proj_wmma_kernel(const float* __restrict__ X,
                 const float* __restrict__ W,
                 const float* __restrict__ bias,
                 float* __restrict__ P,
                 int mTiles) {
  int wave  = blockIdx.x * (blockDim.x >> 5) + (threadIdx.x >> 5);
  int lane  = threadIdx.x & 31;
  int tileN = wave & 3;          // 64 / 16 = 4 N tiles
  int tileM = wave >> 2;
  if (tileM >= mTiles) return;   // wave-uniform: EXEC stays all-ones for WMMA

  int m  = lane & 15;
  int hi = lane >> 4;

  v8f c = {};
#pragma unroll
  for (int k0 = 0; k0 < 64; k0 += 32) {
    v16h af, bf;
    const float* Xrow = X + (size_t)(tileM * 16 + m) * DD + k0;
#pragma unroll
    for (int j = 0; j < 16; ++j) {
      int k = hi * 16 + j;
      af[j] = (_Float16)Xrow[k];
    }
    const float* Wrow = W + (size_t)(tileN * 16 + m) * DD + k0 + hi * 16;
#pragma unroll
    for (int j = 0; j < 16; ++j) bf[j] = (_Float16)Wrow[j];

    c = __builtin_amdgcn_wmma_f32_16x16x32_f16(false, af, false, bf,
                                               (short)0, c, false, false);
  }

  float bn = bias[tileN * 16 + m];           // column = lane&15
#pragma unroll
  for (int r = 0; r < 8; ++r) {
    int row = tileM * 16 + r + hi * 8;
    int col = tileN * 16 + m;
    P[(size_t)row * DD + col] = c[r] + bn;
  }
}

// ---------------------------------------------------------------------------
extern "C" void kernel_launch(void* const* d_in, const int* in_sizes, int n_in,
                              void* d_out, int out_size, void* d_ws, size_t ws_size,
                              hipStream_t stream) {
  const float* user_emb = (const float*)d_in[0];
  const float* item_emb = (const float*)d_in[1];
  const float* W        = (const float*)d_in[2];
  const float* bvec     = (const float*)d_in[3];
  const int*   rows     = (const int*)d_in[4];
  const int*   cols     = (const int*)d_in[5];
  const float* vals     = (const float*)d_in[6];
  const int*   users    = (const int*)d_in[7];
  const int*   items    = (const int*)d_in[8];
  const float* u_his    = (const float*)d_in[9];
  const float* i_his    = (const float*)d_in[10];
  float* out = (float*)d_out;

  int uElems = in_sizes[0];            // U * 64
  int iElems = in_sizes[1];            // I * 64
  int NNZ    = in_sizes[4];
  int B      = in_sizes[7];
  int U      = uElems / DD;
  int ND     = uElems + iElems;        // N * 64

  float* ws   = (float*)d_ws;
  float* acc  = ws;
  float* bufA = ws + (size_t)ND;
  float* bufB = ws + (size_t)2 * ND;
  float* u_on = ws + (size_t)3 * ND;
  float* i_on = u_on + (size_t)B * DD;

  const int tpb = 256;
  int n4  = ND >> 2;
  int eb  = (n4 + tpb - 1) / tpb;

  init_concat_kernel<<<eb, tpb, 0, stream>>>(user_emb, item_emb, acc, bufA, uElems, n4);

  float* cur = bufA;
  float* nxt = bufB;
  for (int l = 0; l < LAYERS; ++l) {
    zero_kernel<<<eb, tpb, 0, stream>>>(nxt, n4);
    spmm_atomic_kernel<<<8192, tpb, 0, stream>>>(rows, cols, vals, cur, nxt, NNZ);
    accum_kernel<<<eb, tpb, 0, stream>>>(acc, nxt, n4);
    float* t = cur; cur = nxt; nxt = t;
  }

  int BD  = B * DD;
  int gth = 2 * BD;
  gather_target_kernel<<<(gth + tpb - 1) / tpb, tpb, 0, stream>>>(
      acc, users, items, u_his, i_his, u_on, i_on, out, B, U);

  int mTiles = B / 16;
  int waves  = mTiles * 4;
  int blocks = (waves * 32 + tpb - 1) / tpb;
  proj_wmma_kernel<<<blocks, tpb, 0, stream>>>(u_on, W, bvec, out, mTiles);
  proj_wmma_kernel<<<blocks, tpb, 0, stream>>>(i_on, W, bvec, out + (size_t)2 * BD, mTiles);
}
